// RecurrentNetwork_28716151341274
// MI455X (gfx1250) — matrix operand order repaired and checked by reading, not compiled, not run on hardware
//
#include <hip/hip_runtime.h>
#include <math.h>
#include <stdint.h>

typedef __bf16 bf16;
typedef __attribute__((ext_vector_type(16))) __bf16 v16bf;
typedef __attribute__((ext_vector_type(8)))  __bf16 v8bf;
typedef __attribute__((ext_vector_type(8)))  float  v8f;
typedef __attribute__((ext_vector_type(4)))  uint32_t v4u;
typedef __attribute__((ext_vector_type(8)))  int32_t  v8i;
typedef __attribute__((ext_vector_type(4)))  int32_t  v4i;

#define SEQ      256
#define BATCH    16
#define EDIM     512
#define HDIM     512
#define G4       2048
#define NROWS    4096      // S*B
#define VOC      32000
#define NCHUNK   125       // 32000 / 256
#define LSTM_WGS 16

// ---------------- WMMA fragment helpers (CDNA5 16x16x32 bf16) ----------------

// A: 16(M) x 32(K) bf16 row-major, leading dim `ld` elems.
__device__ __forceinline__ v16bf load_a_frag(const bf16* A, int ld, int kOff) {
  int l = threadIdx.x & 31;
  int m = l & 15;
  int half = l >> 4;
  const bf16* p = A + (size_t)m * ld + kOff + half * 8;
  v8bf lo = *(const v8bf*)p;
  v8bf hi = *(const v8bf*)(p + 16);
  v16bf r;
#pragma unroll
  for (int i = 0; i < 8; ++i) { r[i] = lo[i]; r[i + 8] = hi[i]; }
  return r;
}

// B: 32(K) x 16(N) bf16 stored [n][k] (column's K run contiguous).
__device__ __forceinline__ v16bf load_b_frag(const bf16* B, int ld, int nBase, int kOff) {
  int l = threadIdx.x & 31;
  int n = l & 15;
  int kb = (l >> 4) * 8;
  const bf16* p = B + (size_t)(nBase + n) * ld + kOff + kb;
  v8bf lo = *(const v8bf*)p;
  v8bf hi = *(const v8bf*)(p + 16);
  v16bf r;
#pragma unroll
  for (int i = 0; i < 8; ++i) { r[i] = lo[i]; r[i + 8] = hi[i]; }
  return r;
}

__device__ __forceinline__ v8f wmma_bf16_f32(v16bf a, v16bf b, v8f c) {
  return __builtin_amdgcn_wmma_f32_16x16x32_bf16(false, a, false, b, (short)0, c,
                                                 false, false);
}

__device__ __forceinline__ float sigf(float x) { return 1.0f / (1.0f + __expf(-x)); }

// ---------------- TDM: 2D tile (tileN rows x tileK elems, bf16) -> LDS ----------------
// D# per CDNA5 ISA ch.8: group0 = {count/flags, lds_addr, global_addr, type=2},
// group1 = {data_size, tensor dims, tile dims, strides}. Higher groups zero (2D tile).
__device__ __forceinline__ void tdm_load_2d_bf16(uint32_t ldsOff, const bf16* gptr,
                                                 uint32_t tileK, uint32_t tileN,
                                                 uint32_t strideElems,
                                                 uint32_t tensorD0, uint32_t tensorD1) {
  uint64_t ga = (uint64_t)(uintptr_t)gptr;
  v4u g0;
  g0[0] = 1u;                                               // count=1, user desc
  g0[1] = ldsOff;                                           // LDS byte address
  g0[2] = (uint32_t)(ga & 0xffffffffu);                     // global_addr[31:0]
  g0[3] = (uint32_t)((ga >> 32) & 0x01ffffffu) | (2u << 30);// addr[56:32] | type=2
  v8i g1;
  g1[0] = (int)(1u << 16);                                  // data_size = 1 (2 bytes)
  g1[1] = (int)((tensorD0 & 0xffffu) << 16);                // tensor_dim0[15:0]
  g1[2] = (int)(((tensorD0 >> 16) & 0xffffu) | ((tensorD1 & 0xffffu) << 16));
  g1[3] = (int)(((tensorD1 >> 16) & 0xffffu) | ((tileK & 0xffffu) << 16)); // tile_dim0
  g1[4] = (int)(tileN & 0xffffu);                           // tile_dim1 (tile_dim2=0)
  g1[5] = (int)strideElems;                                 // tensor_dim0_stride[31:0]
  g1[6] = 0;                                                // stride[47:32] | dim1_stride lo
  g1[7] = 0;
  v4i z4 = {0, 0, 0, 0};
  v8i z8 = {0, 0, 0, 0, 0, 0, 0, 0};
  __builtin_amdgcn_tensor_load_to_lds(g0, g1, z4, z4, z8, 0);
}

// Device-scope grid barrier (16 co-resident persistent blocks).
__device__ __forceinline__ void grid_barrier(int* cnt, int* gen, int nblk) {
  __syncthreads();
  __threadfence();
  if (threadIdx.x == 0) {
    int g = __hip_atomic_load(gen, __ATOMIC_RELAXED, __HIP_MEMORY_SCOPE_AGENT);
    if (__hip_atomic_fetch_add(cnt, 1, __ATOMIC_ACQ_REL, __HIP_MEMORY_SCOPE_AGENT) == nblk - 1) {
      __hip_atomic_store(cnt, 0, __ATOMIC_RELAXED, __HIP_MEMORY_SCOPE_AGENT);
      __hip_atomic_fetch_add(gen, 1, __ATOMIC_RELEASE, __HIP_MEMORY_SCOPE_AGENT);
    } else {
      while (__hip_atomic_load(gen, __ATOMIC_ACQUIRE, __HIP_MEMORY_SCOPE_AGENT) == g) {
        __builtin_amdgcn_s_sleep(1);
      }
    }
  }
  __syncthreads();
  __threadfence();
}

// ---------------- Kernel 0: fp32 -> bf16 weight conversion ----------------
__global__ __launch_bounds__(256) void k_cvt(const float* __restrict__ src,
                                             bf16* __restrict__ dst, int n4) {
  int i = blockIdx.x * 256 + threadIdx.x;
  if (i < n4) {
    float4 f = ((const float4*)src)[i];
    bf16* d = dst + (size_t)i * 4;
    d[0] = (bf16)f.x; d[1] = (bf16)f.y; d[2] = (bf16)f.z; d[3] = (bf16)f.w;
  }
}

// ---------------- Kernel 1: xp = gather(embed) @ W_ih^T + (b_ih + b_hh) ----------------
// Grid (256, 16); block 256 = 8 waves. Tile 16 rows x 128 cols, K in 16 chunks of 32.
__global__ __launch_bounds__(256) void k_xp(const int* __restrict__ tok,
                                            const float* __restrict__ embed,
                                            const bf16* __restrict__ wib,
                                            const float* __restrict__ b_ih,
                                            const float* __restrict__ b_hh,
                                            float* __restrict__ xp) {
  __shared__ __attribute__((aligned(16))) bf16 sA[16 * 32];
  __shared__ __attribute__((aligned(16))) bf16 sB[128 * 32];
  int rowbase = blockIdx.x * 16;
  int colbase = blockIdx.y * 128;
  int t = threadIdx.x;
  int w = t >> 5;
  v8f acc = {0.f, 0.f, 0.f, 0.f, 0.f, 0.f, 0.f, 0.f};

  for (int kc = 0; kc < 16; ++kc) {
    int k0 = kc * 32;
    __syncthreads();
    if (t < 128) {  // A tile: gather + cvt
      int m = t >> 3, kq = (t & 7) * 4;
      int token = tok[rowbase + m];
      float4 f = *(const float4*)(embed + (size_t)token * EDIM + k0 + kq);
      bf16* d = sA + m * 32 + kq;
      d[0] = (bf16)f.x; d[1] = (bf16)f.y; d[2] = (bf16)f.z; d[3] = (bf16)f.w;
    }
    {  // B tile: 128 cols x 32 k, bf16 direct copies (512 v8bf slots)
#pragma unroll
      for (int it = 0; it < 2; ++it) {
        int idx = t + it * 256;
        int n = idx >> 2, kq = (idx & 3) * 8;
        *(v8bf*)(sB + n * 32 + kq) =
            *(const v8bf*)(wib + (size_t)(colbase + n) * EDIM + k0 + kq);
      }
    }
    __syncthreads();
    v16bf a = load_a_frag(sA, 32, 0);
    v16bf b = load_b_frag(sB, 32, w * 16, 0);
    acc = wmma_bf16_f32(a, b, acc);
  }
  int l = t & 31, nn = l & 15, half = l >> 4;
  int col = colbase + w * 16 + nn;
  float bias = b_ih[col] + b_hh[col];
#pragma unroll
  for (int r = 0; r < 8; ++r) {
    int m = r + half * 8;
    xp[(size_t)(rowbase + m) * G4 + col] = acc[r] + bias;
  }
}

// ---------------- Kernel 2: persistent LSTM recurrence ----------------
__global__ __launch_bounds__(256) void k_lstm(const float* __restrict__ W_hh,
                                              const float* __restrict__ xp,
                                              bf16* __restrict__ hbuf,
                                              bf16* __restrict__ hs,
                                              int* __restrict__ bar) {
  extern __shared__ __attribute__((aligned(16))) bf16 sW[];  // [128][512]
  __shared__ __attribute__((aligned(16))) float sG[16 * 128];
  int wg = blockIdx.x;
  int HB = wg * 32;
  int t = threadIdx.x;
  int w = t >> 5;
  int l = t & 31, nn = l & 15, half = l >> 4;

  for (int i4 = t; i4 < 128 * 128; i4 += 256) {
    int r = i4 >> 7;
    int kq = (i4 & 127) * 4;
    int q = r >> 5, j = r & 31;
    float4 f = *(const float4*)(W_hh + (size_t)(q * 512 + HB + j) * HDIM + kq);
    bf16* d = sW + (size_t)r * 512 + kq;
    d[0] = (bf16)f.x; d[1] = (bf16)f.y; d[2] = (bf16)f.z; d[3] = (bf16)f.w;
  }
  for (int i = t; i < 16 * 32; i += 256) {
    int b = i >> 5, j = i & 31;
    hbuf[b * 512 + HB + j] = (bf16)0.0f;
  }
  grid_barrier(bar, bar + 1, LSTM_WGS);

  float creg[2] = {0.f, 0.f};

  for (int s = 0; s < SEQ; ++s) {
    const bf16* hprev = hbuf + (size_t)(s & 1) * 16 * 512;
    bf16* hnext = hbuf + (size_t)((s + 1) & 1) * 16 * 512;

    v8f acc = {0.f, 0.f, 0.f, 0.f, 0.f, 0.f, 0.f, 0.f};
#pragma unroll
    for (int kc = 0; kc < 16; ++kc) {
      v16bf a = load_a_frag(hprev, 512, kc * 32);
      v16bf b = load_b_frag(sW, 512, w * 16, kc * 32);
      acc = wmma_bf16_f32(a, b, acc);
    }
#pragma unroll
    for (int r = 0; r < 8; ++r) sG[(r + half * 8) * 128 + w * 16 + nn] = acc[r];
    __syncthreads();

#pragma unroll
    for (int pp = 0; pp < 2; ++pp) {
      int p = t + pp * 256;
      int b = p >> 5, j = p & 31;
      size_t xr = (size_t)(s * 16 + b) * G4;
      float gi = sG[b * 128 + 0  + j] + xp[xr + 0 * 512 + HB + j];
      float gf = sG[b * 128 + 32 + j] + xp[xr + 1 * 512 + HB + j];
      float gg = sG[b * 128 + 64 + j] + xp[xr + 2 * 512 + HB + j];
      float go = sG[b * 128 + 96 + j] + xp[xr + 3 * 512 + HB + j];
      float cn = sigf(gf) * creg[pp] + sigf(gi) * tanhf(gg);
      float h  = sigf(go) * tanhf(cn);
      creg[pp] = cn;
      hnext[b * 512 + HB + j] = (bf16)h;
      hs[(size_t)(s * 16 + b) * HDIM + HB + j] = (bf16)h;
    }
    grid_barrier(bar, bar + 1, LSTM_WGS);
  }
}

// ---------------- Kernel 3: logits GEMM (TDM B tiles) + per-chunk softmax stats ----------------
// Grid (128, 125); block 256 = 8 waves. Tile 32 rows x 256 cols; K in 16 chunks of 32.
// B tiles DMA'd via tensor_load_to_lds, double buffered, TENSORcnt-synced.
__global__ __launch_bounds__(256) void k_logits(const bf16* __restrict__ hs,
                                                const bf16* __restrict__ wlb,
                                                const float* __restrict__ b_lin,
                                                float* __restrict__ out,
                                                float* __restrict__ stats) {
  __shared__ __attribute__((aligned(16))) bf16 sB[2][256 * 32];  // 2 x 16KB
  __shared__ __attribute__((aligned(16))) bf16 sA[32 * 32];      // 2KB
  __shared__ float sRed[256];
  __shared__ float sMax[32];
  float* sL = (float*)&sB[0][0];  // 32KB alias, used after K loop only

  int rowbase = blockIdx.x * 32;
  int colbase = blockIdx.y * 256;
  int t = threadIdx.x;
  int w = t >> 5;
  v8f acc[4];
#pragma unroll
  for (int i = 0; i < 4; ++i) acc[i] = (v8f){0.f, 0.f, 0.f, 0.f, 0.f, 0.f, 0.f, 0.f};

  const bf16* bsrc = wlb + (size_t)colbase * HDIM;

  // Prologue: DMA chunk 0 into buffer 0 (one issue per block; TDM ignores EXEC).
  if (t < 32) {
    tdm_load_2d_bf16((uint32_t)(uintptr_t)&sB[0][0], bsrc, 32, 256, HDIM, HDIM, VOC);
  }

  for (int kc = 0; kc < 16; ++kc) {
    int k0 = kc * 32;
    __syncthreads();  // previous chunk's LDS reads complete
    if (t < 128) {    // A tile: 32 rows x 32 k bf16 (128 v8bf slots)
      int m = t >> 2, kq = (t & 3) * 8;
      *(v8bf*)(sA + m * 32 + kq) =
          *(const v8bf*)(hs + (size_t)(rowbase + m) * HDIM + k0 + kq);
    }
    if (t < 32) {
      if (kc + 1 < 16) {
        tdm_load_2d_bf16((uint32_t)(uintptr_t)&sB[(kc + 1) & 1][0],
                         bsrc + (kc + 1) * 32, 32, 256, HDIM, HDIM, VOC);
        __builtin_amdgcn_s_wait_tensorcnt(1);  // oldest (chunk kc) done
      } else {
        __builtin_amdgcn_s_wait_tensorcnt(0);
      }
    }
    __syncthreads();  // publish TDM data + sA to all waves
    const bf16* bb = &sB[kc & 1][0];
    v16bf a0 = load_a_frag(sA, 32, 0);
    v16bf a1 = load_a_frag(sA + 16 * 32, 32, 0);
    v16bf b0 = load_b_frag(bb, 32, w * 32, 0);
    v16bf b1 = load_b_frag(bb, 32, w * 32 + 16, 0);
    acc[0] = wmma_bf16_f32(a0, b0, acc[0]);
    acc[1] = wmma_bf16_f32(a0, b1, acc[1]);
    acc[2] = wmma_bf16_f32(a1, b0, acc[2]);
    acc[3] = wmma_bf16_f32(a1, b1, acc[3]);
  }
  __syncthreads();  // last WMMA reads done before aliasing sB as sL
  int l = t & 31, nn = l & 15, half = l >> 4;
#pragma unroll
  for (int mt2 = 0; mt2 < 2; ++mt2) {
#pragma unroll
    for (int nt2 = 0; nt2 < 2; ++nt2) {
      v8f a = acc[mt2 * 2 + nt2];
      int col = w * 32 + nt2 * 16 + nn;
      float bl = b_lin[colbase + col];
#pragma unroll
      for (int r = 0; r < 8; ++r) {
        int m = mt2 * 16 + r + half * 8;
        sL[m * 256 + col] = a[r] + bl;
      }
    }
  }
  __syncthreads();
  {  // write raw logits: 32 rows x 256 cols
    int m = t >> 3, seg = t & 7;
    const float* src = sL + m * 256;
    float* dst = out + (size_t)(rowbase + m) * VOC + colbase;
    for (int c4 = seg; c4 < 64; c4 += 8)
      *(float4*)(dst + c4 * 4) = *(const float4*)(src + c4 * 4);
  }
  // chunk-local max, then sum(exp(x - max)) per row (8 threads x 32 cols per row)
  int m = t >> 3, seg = t & 7;
  float mx = -1e30f;
#pragma unroll
  for (int c = 0; c < 32; ++c) mx = fmaxf(mx, sL[m * 256 + seg * 32 + c]);
  sRed[t] = mx;
  __syncthreads();
  if (t < 32) {
    float v = -1e30f;
    for (int i = 0; i < 8; ++i) v = fmaxf(v, sRed[t * 8 + i]);
    sMax[t] = v;
  }
  __syncthreads();
  float rm = sMax[m];
  float sum = 0.f;
#pragma unroll
  for (int c = 0; c < 32; ++c) sum += __expf(sL[m * 256 + seg * 32 + c] - rm);
  sRed[t] = sum;
  __syncthreads();
  if (t < 32) {
    float s2 = 0.f;
    for (int i = 0; i < 8; ++i) s2 += sRed[t * 8 + i];
    size_t o = ((size_t)(rowbase + t) * NCHUNK + blockIdx.y) * 2;
    stats[o] = sMax[t];
    stats[o + 1] = s2;
  }
}

// ---------------- Kernel 4: merge chunk stats per row ----------------
__global__ __launch_bounds__(256) void k_reduce(const float* __restrict__ stats,
                                                float* __restrict__ fin) {
  int r = blockIdx.x * 256 + threadIdx.x;
  float m = -1e30f, lsum = 0.f;
  for (int c = 0; c < NCHUNK; ++c) {
    float mc = stats[((size_t)r * NCHUNK + c) * 2];
    float lc = stats[((size_t)r * NCHUNK + c) * 2 + 1];
    if (mc > m) { lsum = lsum * __expf(m - mc) + lc; m = mc; }
    else        { lsum += lc * __expf(mc - m); }
  }
  fin[r * 2] = m;
  fin[r * 2 + 1] = logf(lsum);
}

// ---------------- Kernel 5: out = logits - (max + log(sumexp)) ----------------
__global__ __launch_bounds__(256) void k_fixup(float* __restrict__ out,
                                               const float* __restrict__ fin) {
  int row = blockIdx.x;
  float sub = fin[row * 2] + fin[row * 2 + 1];
  float4* p = (float4*)(out + (size_t)row * VOC);
  for (int i = threadIdx.x; i < VOC / 4; i += 256) {
    float4 v = p[i];
    v.x -= sub; v.y -= sub; v.z -= sub; v.w -= sub;
    p[i] = v;
  }
}

// ---------------- launch ----------------
extern "C" void kernel_launch(void* const* d_in, const int* in_sizes, int n_in,
                              void* d_out, int out_size, void* d_ws, size_t ws_size,
                              hipStream_t stream) {
  const int*   tok    = (const int*)d_in[0];
  // d_in[1] = input_lengths (unused by the reference math)
  const float* embed  = (const float*)d_in[2];
  const float* W_ih   = (const float*)d_in[3];
  const float* W_hh   = (const float*)d_in[4];
  const float* b_ih   = (const float*)d_in[5];
  const float* b_hh   = (const float*)d_in[6];
  const float* W_lin  = (const float*)d_in[7];
  const float* b_lin  = (const float*)d_in[8];
  float* out = (float*)d_out;

  char* ws = (char*)d_ws;
  size_t o = 0;
  float* xp    = (float*)(ws + o); o += (size_t)NROWS * G4 * 4;         // 32 MB
  bf16*  hs    = (bf16*)(ws + o);  o += (size_t)NROWS * HDIM * 2;       // 4 MB
  bf16*  hbuf  = (bf16*)(ws + o);  o += (size_t)2 * 16 * 512 * 2;       // 32 KB
  float* stats = (float*)(ws + o); o += (size_t)NROWS * NCHUNK * 2 * 4; // ~3.9 MB
  float* fin   = (float*)(ws + o); o += (size_t)NROWS * 2 * 4;          // 32 KB
  int*   bar   = (int*)(ws + o);   o += 256;
  bf16*  wlb   = (bf16*)(ws + o);  o += (size_t)VOC * HDIM * 2;         // 31.25 MB
  bf16*  wib   = (bf16*)(ws + o);  o += (size_t)G4 * EDIM * 2;          // 2 MB

  (void)hipMemsetAsync(bar, 0, 256, stream);

  // Pre-convert weights to bf16 (amortized over all tiles; stays L2-resident).
  k_cvt<<<(VOC * HDIM / 4 + 255) / 256, 256, 0, stream>>>(W_lin, wlb, VOC * HDIM / 4);
  k_cvt<<<(G4 * EDIM / 4 + 255) / 256, 256, 0, stream>>>(W_ih, wib, G4 * EDIM / 4);

  k_xp<<<dim3(256, 16), 256, 0, stream>>>(tok, embed, wib, b_ih, b_hh, xp);

  size_t lstmSmem = (size_t)128 * 512 * sizeof(bf16);  // 128 KB dynamic LDS
  (void)hipFuncSetAttribute((const void*)k_lstm,
                            hipFuncAttributeMaxDynamicSharedMemorySize,
                            (int)lstmSmem);
  k_lstm<<<LSTM_WGS, 256, lstmSmem, stream>>>(W_hh, xp, hbuf, hs, bar);

  k_logits<<<dim3(128, NCHUNK), 256, 0, stream>>>(hs, wlb, b_lin, out, stats);
  k_reduce<<<NROWS / 256, 256, 0, stream>>>(stats, fin);
  k_fixup<<<NROWS, 256, 0, stream>>>(out, fin);
}
